// Attention_48799418417201
// MI455X (gfx1250) — matrix operand order repaired and checked by reading, not compiled, not run on hardware
//
#include <hip/hip_runtime.h>

// ---------------------------------------------------------------------------
// MI455X (gfx1250) LoRA-MHA, bf16 WMMA pipeline + CDNA5 async/TDM staging.
// Shapes: B=8, S=1024, E=768, H=12, hd=64, R=16.  M = B*S = 8192.
// LoRA folded into weights: W_eff = in_proj_weight + lora_b @ lora_a.
// Matmuls: v_wmma_f32_16x16x32_bf16 (wave32), f32 accumulate.
// LDS staging: global_load_async_to_lds_b128 (ASYNCcnt) in the GEMMs,
// tensor_load_to_lds (TDM, TENSORcnt) for the attention K tiles.
// ---------------------------------------------------------------------------

typedef __attribute__((ext_vector_type(16))) __bf16 v16bf;
typedef __attribute__((ext_vector_type(8)))  __bf16 v8bf;
typedef __attribute__((ext_vector_type(4)))  __bf16 v4bf;
typedef __attribute__((ext_vector_type(8)))  float  v8f;
typedef __attribute__((ext_vector_type(4)))  unsigned v4u;
typedef __attribute__((ext_vector_type(8)))  unsigned v8u;

#define E_DIM   768
#define M_TOT   8192
#define S_LEN   1024
#define HD      64
#define NH      12
#define NB      8
#define LDA     40      // LDS row stride (bf16 elems), 80B -> 16B aligned rows
#define LDB     40
#define LDK     72      // attention LDS stride (144B rows)
#define SLICE   (M_TOT*E_DIM)        // 6291456 elems per q/k/v slice
#define BH_STR  (S_LEN*HD)           // 65536 elems per (b,h) matrix

// Low 32 bits of a flat pointer into LDS == LDS byte offset (ISA 10.2).
__device__ __forceinline__ unsigned lds_addr32(const void* p) {
  return (unsigned)(uintptr_t)p;
}

// Async copy of 16B memory -> LDS, per lane. IOFFSET applies to both sides.
__device__ __forceinline__ void async_b128(unsigned lds, const void* gaddr) {
  asm volatile("global_load_async_to_lds_b128 %0, %1, off"
               :: "v"(lds), "v"(gaddr) : "memory");
}
__device__ __forceinline__ void async_b128_off16(unsigned lds, const void* gaddr) {
  asm volatile("global_load_async_to_lds_b128 %0, %1, off offset:16"
               :: "v"(lds), "v"(gaddr) : "memory");
}
__device__ __forceinline__ void wait_async0() {
  asm volatile("s_wait_asynccnt 0x0" ::: "memory");
}

// Combine two 16B LDS/global chunks into one 16-element bf16 fragment.
__device__ __forceinline__ v16bf ld16(const __bf16* p0, const __bf16* p1) {
  v8bf lo = *(const v8bf*)p0;
  v8bf hi = *(const v8bf*)p1;
  return __builtin_shufflevector(lo, hi, 0,1,2,3,4,5,6,7,8,9,10,11,12,13,14,15);
}

// ---------------------------------------------------------------------------
// Kernel 1: W_eff[j][k] = in_proj_weight[j][k] + sum_r lora_b[j][r]*lora_a[r][k]
// stored bf16 row-major [n][k] (a WMMA B-column == contiguous weight row).
// ---------------------------------------------------------------------------
__global__ __launch_bounds__(256) void prep_w(const float* __restrict__ ipw,
                                              const float* __restrict__ la,
                                              const float* __restrict__ lb,
                                              __bf16* __restrict__ Wbf) {
  const int idx = blockIdx.x * 256 + threadIdx.x;      // < 2304*768
  const int j = idx / E_DIM, k = idx - j * E_DIM;
  float acc = ipw[idx];
#pragma unroll
  for (int r = 0; r < 16; ++r)
    acc = fmaf(lb[j * 16 + r], la[r * E_DIM + k], acc);
  Wbf[idx] = (__bf16)acc;
}

// f32 -> bf16, 4 elements per thread
__global__ __launch_bounds__(256) void cvt4(const float* __restrict__ src,
                                            __bf16* __restrict__ dst, int n4) {
  const int i = blockIdx.x * 256 + threadIdx.x;
  if (i >= n4) return;
  const float4 v = ((const float4*)src)[i];
  v4bf o;
  o[0] = (__bf16)v.x; o[1] = (__bf16)v.y; o[2] = (__bf16)v.z; o[3] = (__bf16)v.w;
  *(v4bf*)(dst + i * 4) = o;
}

// ---------------------------------------------------------------------------
// Shared GEMM core: C(128x64) = X(128xK) * W^T, W stored [n][k].
// Block = 8 waves; wave (w&3, w>>2) owns a 32x32 patch = 2x2 WMMA tiles.
// LDS staged with global_load_async_to_lds_b128 (no VGPR round trip).
// Fragment layouts follow CDNA5 ISA 7.12.2 exactly (wave32).
// ---------------------------------------------------------------------------
__device__ __forceinline__ void gemm_core(const __bf16* __restrict__ X,
                                          const __bf16* __restrict__ W,
                                          int m0, int n0, int K,
                                          __bf16* ldsA, __bf16* ldsB,
                                          v8f acc[2][2]) {
  const int tid  = threadIdx.x;
  const int w    = tid >> 5;
  const int l    = tid & 31;
  const int lr   = l & 15;
  const int half = l >> 4;
  const int wm   = (w & 3) * 32;
  const int wn   = (w >> 2) * 32;
  const int ar = tid >> 1, ak = (tid & 1) * 16;   // A stage: 128 rows x 32 k
  const int br = tid >> 2, bk = (tid & 3) * 8;    // B stage:  64 rows x 32 k
  const unsigned ldsAaddr = lds_addr32(&ldsA[ar * LDA + ak]);
  const unsigned ldsBaddr = lds_addr32(&ldsB[br * LDB + bk]);

#pragma unroll
  for (int i = 0; i < 2; ++i)
#pragma unroll
    for (int j = 0; j < 2; ++j)
#pragma unroll
      for (int e = 0; e < 8; ++e) acc[i][j][e] = 0.f;

  for (int k0 = 0; k0 < K; k0 += 32) {
    __syncthreads();                       // prev fragments consumed
    {
      const __bf16* ap = X + (m0 + ar) * K + k0 + ak;     // 32B contiguous
      async_b128(ldsAaddr, ap);
      async_b128_off16(ldsAaddr, ap);
      const __bf16* bp = W + (n0 + br) * K + k0 + bk;     // 16B contiguous
      async_b128(ldsBaddr, bp);
      if (k0 + 32 < K) {                   // global_prefetch_b8 for next tile
        __builtin_prefetch(ap + 32, 0, 0);
        __builtin_prefetch(bp + 32, 0, 0);
      }
    }
    wait_async0();                         // our async stores landed in LDS
    __syncthreads();                       // everyone's did

    v16bf afr[2], bfr[2];
#pragma unroll
    for (int i = 0; i < 2; ++i) {
      // A 16x32: lane half-split K runs of 8 (K = half*8+i, 16+half*8+i)
      const __bf16* pa = &ldsA[(wm + i * 16 + lr) * LDA + half * 8];
      afr[i] = ld16(pa, pa + 16);
    }
#pragma unroll
    for (int j = 0; j < 2; ++j) {
      // B 32x16: lane = column N, K = half*8+i, 16+half*8+i (matches A)
      const __bf16* pb = &ldsB[(wn + j * 16 + lr) * LDB + half * 8];
      bfr[j] = ld16(pb, pb + 16);
    }
#pragma unroll
    for (int i = 0; i < 2; ++i)
#pragma unroll
      for (int j = 0; j < 2; ++j)
        acc[i][j] = __builtin_amdgcn_wmma_f32_16x16x32_bf16(
            false, afr[i], false, bfr[j], (short)0, acc[i][j], false, false);
  }
}

// ---------------------------------------------------------------------------
// Kernel 2: QKV projection. grid = (64, 12, 3). Output scattered into
// head layout QKVh[s][b][h][srow][d], q pre-scaled by 1/sqrt(hd)=0.125.
// ---------------------------------------------------------------------------
__global__ __launch_bounds__(256) void proj_gemm(const __bf16* __restrict__ Xbf,
                                                 const __bf16* __restrict__ Wbf,
                                                 const float* __restrict__ bias,
                                                 __bf16* __restrict__ QKVh) {
  __shared__ __bf16 ldsA[128 * LDA];
  __shared__ __bf16 ldsB[64 * LDB];
  const int s  = blockIdx.z;
  const int m0 = blockIdx.x * 128, n0 = blockIdx.y * 64;
  const __bf16* X = Xbf + s * SLICE;
  const __bf16* W = Wbf + s * (E_DIM * E_DIM);
  const float* bb = bias + s * E_DIM;
  const float scale = (s == 0) ? 0.125f : 1.0f;

  v8f acc[2][2];
  gemm_core(X, W, m0, n0, E_DIM, ldsA, ldsB, acc);

  const int l = threadIdx.x & 31, lr = l & 15, half = l >> 4;
  const int w = threadIdx.x >> 5;
  const int wm = (w & 3) * 32, wn = (w >> 2) * 32;
  __bf16* base = QKVh + s * SLICE;
#pragma unroll
  for (int i = 0; i < 2; ++i)
#pragma unroll
    for (int j = 0; j < 2; ++j) {
      const int gn = n0 + wn + j * 16 + lr;
      const int h = gn >> 6, d = gn & 63;
      const float bv = bb[gn];
#pragma unroll
      for (int e = 0; e < 8; ++e) {
        const int gm = m0 + wm + i * 16 + half * 8 + e;     // C/D: M = e + half*8
        const int bi = gm >> 10, srow = gm & 1023;
        base[((bi * NH + h) << 16) + srow * HD + d] =
            (__bf16)((acc[i][j][e] + bv) * scale);
      }
    }
}

// ---------------------------------------------------------------------------
// Kernel 3: flash attention. grid = (S/128, H, B), 8 waves, wave w owns
// 16 full query rows -> online softmax stats stay inside one lane-half.
// K tile (one contiguous 8KB run) is pulled by the Tensor Data Mover.
// ---------------------------------------------------------------------------
__global__ __launch_bounds__(256) void flash_attn(const __bf16* __restrict__ QKVh,
                                                  __bf16* __restrict__ AO) {
  const int b = blockIdx.z, h = blockIdx.y, q0 = blockIdx.x * 128;
  const int tid = threadIdx.x, w = tid >> 5, l = tid & 31;
  const int lr = l & 15, half = l >> 4;
  const __bf16* Qb = QKVh + (b * NH + h) * BH_STR;
  const __bf16* Kb = Qb + NB * NH * BH_STR;        // slice 1
  const __bf16* Vb = Qb + 2 * NB * NH * BH_STR;    // slice 2

  __shared__ __bf16 Kt[64 * 64];      // [key][d] (dense 128B rows, TDM target)
  __shared__ __bf16 VT[64 * LDK];     // [d][key]  (transposed at load)
  __shared__ __bf16 Pl[8 * 16 * LDK]; // per-wave P patch [row][key]

  // Q A-fragments (rows q0+16w+lr), 2 K-steps over hd=64
  v16bf aq[2];
#pragma unroll
  for (int ks = 0; ks < 2; ++ks) {
    const __bf16* p = Qb + (q0 + w * 16 + lr) * HD + ks * 32 + half * 8;
    aq[ks] = ld16(p, p + 16);
  }

  v8f m, lsum, oacc[4];
#pragma unroll
  for (int e = 0; e < 8; ++e) { m[e] = -3.0e38f; lsum[e] = 0.f; }
#pragma unroll
  for (int dt = 0; dt < 4; ++dt)
#pragma unroll
    for (int e = 0; e < 8; ++e) oacc[dt][e] = 0.f;

  __bf16* pw = Pl + w * 16 * LDK;

  for (int kt = 0; kt < 16; ++kt) {
    __syncthreads();
    // --- K tile via Tensor Data Mover: 64x128B contiguous = 8192B, 1 op ---
    if (w == 0) {
      const unsigned long long ga = (unsigned long long)(uintptr_t)(Kb + kt * 64 * HD);
      v4u g0;                                     // D# group 0 (ISA 8.3)
      g0[0] = 1u;                                 // count=1, user descriptor
      g0[1] = lds_addr32(&Kt[0]);                 // lds_addr
      g0[2] = (unsigned)ga;                       // global_addr[31:0]
      g0[3] = (unsigned)((ga >> 32) & 0x01FFFFFFu) | (2u << 30);  // type=2
      v8u g1;                                     // D# group 1 (ISA 8.4)
      g1[0] = 3u << 16;                           // data_size = 8B units
      g1[1] = 0u;                                 // tensor_dim0[15:0]=0 (of 1M)
      g1[2] = 0x10u | (1u << 16);                 // tensor_dim0 hi, tensor_dim1=1
      g1[3] = 1024u << 16;                        // tile_dim0 = 1024 x 8B
      g1[4] = 1u;                                 // tile_dim1 = 1
      g1[5] = 0x100000u;                          // tensor_dim0_stride = 1M
      g1[6] = 0u; g1[7] = 0u;
      asm volatile("tensor_load_to_lds %0, %1" :: "s"(g0), "s"(g1) : "memory");
    }
    {   // cooperative stage: V tile transposed (manual; TDM cannot transpose)
      const int key = tid >> 2, doff = (tid & 3) * 16;
      const __bf16* vp = Vb + (kt * 64 + key) * HD + doff;
      v8bf v0 = *(const v8bf*)vp, v1 = *(const v8bf*)(vp + 8);
#pragma unroll
      for (int i = 0; i < 8; ++i) {
        VT[(doff + i) * LDK + key]     = v0[i];
        VT[(doff + 8 + i) * LDK + key] = v1[i];
      }
    }
    if (w == 0) __builtin_amdgcn_s_wait_tensorcnt(0);
    __syncthreads();

    // S = Q * K^T   (16 x 64 per wave)
    v8f sc[4];
#pragma unroll
    for (int nt = 0; nt < 4; ++nt) {
#pragma unroll
      for (int e = 0; e < 8; ++e) sc[nt][e] = 0.f;
#pragma unroll
      for (int ks = 0; ks < 2; ++ks) {
        const __bf16* pb = &Kt[(nt * 16 + lr) * HD + ks * 32 + half * 8];
        v16bf bk = ld16(pb, pb + 16);
        sc[nt] = __builtin_amdgcn_wmma_f32_16x16x32_bf16(
            false, aq[ks], false, bk, (short)0, sc[nt], false, false);
      }
    }

    // online softmax: reduce across the 16 lanes of this half
    v8f rm;
#pragma unroll
    for (int e = 0; e < 8; ++e) {
      float x = fmaxf(fmaxf(sc[0][e], sc[1][e]), fmaxf(sc[2][e], sc[3][e]));
      x = fmaxf(x, __shfl_xor(x, 1, 32));
      x = fmaxf(x, __shfl_xor(x, 2, 32));
      x = fmaxf(x, __shfl_xor(x, 4, 32));
      x = fmaxf(x, __shfl_xor(x, 8, 32));
      rm[e] = x;
    }
    v8f mnew, alpha, rs;
#pragma unroll
    for (int e = 0; e < 8; ++e) {
      mnew[e]  = fmaxf(m[e], rm[e]);
      alpha[e] = __expf(m[e] - mnew[e]);
      rs[e] = 0.f;
    }
#pragma unroll
    for (int nt = 0; nt < 4; ++nt)
#pragma unroll
      for (int e = 0; e < 8; ++e) {
        const float p = __expf(sc[nt][e] - mnew[e]);
        rs[e] += p;
        pw[(e + half * 8) * LDK + nt * 16 + lr] = (__bf16)p;   // C-layout -> LDS
      }
#pragma unroll
    for (int e = 0; e < 8; ++e) {
      float x = rs[e];
      x += __shfl_xor(x, 1, 32);
      x += __shfl_xor(x, 2, 32);
      x += __shfl_xor(x, 4, 32);
      x += __shfl_xor(x, 8, 32);
      lsum[e] = lsum[e] * alpha[e] + x;
      m[e] = mnew[e];
    }
#pragma unroll
    for (int dt = 0; dt < 4; ++dt)
#pragma unroll
      for (int e = 0; e < 8; ++e) oacc[dt][e] *= alpha[e];
    __syncthreads();   // P visible, A-layout reload next

    // O += P * V
    v16bf ap[2];
#pragma unroll
    for (int ks = 0; ks < 2; ++ks) {
      const __bf16* pp = pw + lr * LDK + ks * 32 + half * 8;
      ap[ks] = ld16(pp, pp + 16);
    }
#pragma unroll
    for (int dt = 0; dt < 4; ++dt)
#pragma unroll
      for (int ks = 0; ks < 2; ++ks) {
        const __bf16* pv = &VT[(dt * 16 + lr) * LDK + ks * 32 + half * 8];
        v16bf bv = ld16(pv, pv + 16);
        oacc[dt] = __builtin_amdgcn_wmma_f32_16x16x32_bf16(
            false, ap[ks], false, bv, (short)0, oacc[dt], false, false);
      }
  }

  // epilogue: O /= l ; write AO[b][row][h*64+d]
#pragma unroll
  for (int dt = 0; dt < 4; ++dt) {
    const int d = dt * 16 + lr;
#pragma unroll
    for (int e = 0; e < 8; ++e) {
      const int row = q0 + w * 16 + half * 8 + e;
      AO[(b * S_LEN + row) * E_DIM + h * HD + d] =
          (__bf16)(oacc[dt][e] * (1.0f / lsum[e]));
    }
  }
}

// ---------------------------------------------------------------------------
// Kernel 4: out projection -> f32 d_out. grid = (64, 12)
// ---------------------------------------------------------------------------
__global__ __launch_bounds__(256) void out_gemm(const __bf16* __restrict__ AO,
                                                const __bf16* __restrict__ OW,
                                                const float* __restrict__ ob,
                                                float* __restrict__ out) {
  __shared__ __bf16 ldsA[128 * LDA];
  __shared__ __bf16 ldsB[64 * LDB];
  const int m0 = blockIdx.x * 128, n0 = blockIdx.y * 64;
  v8f acc[2][2];
  gemm_core(AO, OW, m0, n0, E_DIM, ldsA, ldsB, acc);

  const int l = threadIdx.x & 31, lr = l & 15, half = l >> 4;
  const int w = threadIdx.x >> 5;
  const int wm = (w & 3) * 32, wn = (w >> 2) * 32;
#pragma unroll
  for (int i = 0; i < 2; ++i)
#pragma unroll
    for (int j = 0; j < 2; ++j) {
      const int gn = n0 + wn + j * 16 + lr;
      const float bv = ob[gn];
#pragma unroll
      for (int e = 0; e < 8; ++e) {
        const int gm = m0 + wm + i * 16 + half * 8 + e;
        out[gm * E_DIM + gn] = acc[i][j][e] + bv;
      }
    }
}

// ---------------------------------------------------------------------------
extern "C" void kernel_launch(void* const* d_in, const int* in_sizes, int n_in,
                              void* d_out, int out_size, void* d_ws, size_t ws_size,
                              hipStream_t stream) {
  const float* q   = (const float*)d_in[0];
  const float* k   = (const float*)d_in[1];
  const float* v   = (const float*)d_in[2];
  const float* ipw = (const float*)d_in[3];
  const float* ipb = (const float*)d_in[4];
  const float* ow  = (const float*)d_in[5];
  const float* ob  = (const float*)d_in[6];
  const float* la  = (const float*)d_in[7];
  const float* lb  = (const float*)d_in[8];
  (void)in_sizes; (void)n_in; (void)out_size; (void)ws_size;

  __bf16* ws   = (__bf16*)d_ws;
  __bf16* Wbf  = ws;                       // 3*768*768       = 1,769,472
  __bf16* OWbf = ws + 1769472;             //   768*768       =   589,824
  __bf16* Xbf  = ws + 2359296;             // 3*8192*768      = 18,874,368
  __bf16* QKVh = ws + 21233664;            // 3*8192*768 (head layout)
  __bf16* AO   = ws + 40108032;            //   8192*768

  prep_w<<<6912, 256, 0, stream>>>(ipw, la, lb, Wbf);
  cvt4<<<6144, 256, 0, stream>>>(q, Xbf,               SLICE / 4);
  cvt4<<<6144, 256, 0, stream>>>(k, Xbf + SLICE,       SLICE / 4);
  cvt4<<<6144, 256, 0, stream>>>(v, Xbf + 2 * SLICE,   SLICE / 4);
  cvt4<<<576,  256, 0, stream>>>(ow, OWbf,             (E_DIM * E_DIM) / 4);
  proj_gemm<<<dim3(64, 12, 3), 256, 0, stream>>>(Xbf, Wbf, ipb, QKVh);
  flash_attn<<<dim3(8, NH, NB), 256, 0, stream>>>(QKVh, AO);
  out_gemm<<<dim3(64, 12, 1), 256, 0, stream>>>(AO, OWbf, ob, (float*)d_out);
}
